// SubcarrierCrossAttn_Band_36309653520514
// MI455X (gfx1250) — matrix operand order repaired and checked by reading, not compiled, not run on hardware
//
#include <hip/hip_runtime.h>
#include <hip/hip_bf16.h>
#include <math.h>

typedef __attribute__((ext_vector_type(16))) _Float16 v16h;
typedef __attribute__((ext_vector_type(8)))  _Float16 v8h;
typedef __attribute__((ext_vector_type(8)))  float    v8f;

#define BAND 45
#define DM   64
#define NH   4
#define NA   6
#define HD   16
#define KSUB 8          // 360 / 45
#define TOK  8          // tokens per workgroup
#define ROWS (TOK * NA) // 48 rows
#define BLOCK 96        // 3 waves, one 16-row M-tile each

// f16 weight buffer layout inside d_ws (all row-major [n][64], zero-padded)
#define EMB_OFF  0              // embed_w   [64][64]  (k 45..63 = 0)
#define INP_OFF  4096           // in_proj_w [192][64]
#define OUTP_OFF 16384          // out_proj_w[64][64]
#define OUTW_OFF 20480          // out_w     [48][64]  (rows 45..47 = 0)
#define WBUF_ELEMS 23552

struct __align__(16) SMem {
    _Float16 X[ROWS][64];     // padded input (k 45..63 = 0), f16
    _Float16 h16[ROWS][64];   // embed output, f16 (A operand for in_proj)
    float    h32[ROWS][64];   // embed output f32 (residual); later pre-LN sum
    _Float16 qkv[ROWS][192];  // q | k | v
    _Float16 o16[ROWS][64];   // attention output (A operand for out_proj)
    _Float16 y16[ROWS][64];   // LayerNorm output (A operand for head GEMM)
};

// A-fragment (16x32 f16) from row-major f16 LDS array with 64-half row stride.
__device__ __forceinline__ v16h load_a_frag(const _Float16* base, int mbase, int ks, int lane) {
    int row = mbase + (lane & 15);
    int sel = lane >> 4;
    const _Float16* p = base + row * 64 + ks * 32 + sel * 16;
    union { v16h v; v8h h[2]; } u;
    u.h[0] = *(const v8h*)(p);
    u.h[1] = *(const v8h*)(p + 8);
    return u.v;
}

// B-fragment (32x16 f16) from pre-converted f16 weights, row-major [n][64].
// lanes 0-15: n=lane, k = ks*32 + 0..15 ; lanes 16-31: n=lane-16, k = ks*32 + 16..31
__device__ __forceinline__ v16h load_b_frag16(const _Float16* __restrict__ W16,
                                              int ntile, int ks, int lane) {
    int n  = ntile * 16 + (lane & 15);
    int k0 = ks * 32 + (lane >> 4) * 16;
    const _Float16* p = W16 + n * 64 + k0;
    union { v16h v; v8h h[2]; } u;
    u.h[0] = *(const v8h*)(p);
    u.h[1] = *(const v8h*)(p + 8);
    return u.v;
}

// Fallback: B-fragment straight from global row-major f32 weight W[nrows][ldk].
__device__ __forceinline__ v16h load_b_frag32(const float* __restrict__ W, int ldk, int nrows,
                                              int ntile, int ks, int lane) {
    int n  = ntile * 16 + (lane & 15);
    int k0 = ks * 32 + (lane >> 4) * 16;
    v16h b;
#pragma unroll
    for (int i = 0; i < 16; ++i) {
        int k = k0 + i;
        float w = (n < nrows && k < ldk) ? W[n * ldk + k] : 0.0f;
        b[i] = (_Float16)w;
    }
    return b;
}

__device__ __forceinline__ v8f wmma_f16(v16h a, v16h b, v8f c) {
    return __builtin_amdgcn_wmma_f32_16x16x32_f16(false, a, false, b, (short)0, c, false, false);
}

// ---- one-shot weight conversion: fp32 -> padded f16 in d_ws ----
__global__ void convert_weights_kernel(const float* __restrict__ embed_w,
                                       const float* __restrict__ in_proj_w,
                                       const float* __restrict__ out_proj_w,
                                       const float* __restrict__ out_w,
                                       _Float16* __restrict__ wbuf) {
    int i = blockIdx.x * blockDim.x + threadIdx.x;
    if (i >= WBUF_ELEMS) return;
    float v;
    if (i < INP_OFF) {                       // embed_w [64][45] -> [64][64]
        int r = i >> 6, c = i & 63;
        v = (c < BAND) ? embed_w[r * BAND + c] : 0.0f;
    } else if (i < OUTP_OFF) {               // in_proj_w [192][64]
        v = in_proj_w[i - INP_OFF];
    } else if (i < OUTW_OFF) {               // out_proj_w [64][64]
        v = out_proj_w[i - OUTP_OFF];
    } else {                                 // out_w [45][64] -> [48][64]
        int j = i - OUTW_OFF;
        int r = j >> 6, c = j & 63;
        v = (r < BAND) ? out_w[r * DM + c] : 0.0f;
    }
    wbuf[i] = (_Float16)v;
}

template <bool W16F>
__global__ __launch_bounds__(BLOCK)
void subcarrier_cross_attn_kernel(
    const float* __restrict__ x,
    const float* __restrict__ embed_w,  const float* __restrict__ embed_b,
    const float* __restrict__ in_proj_w, const float* __restrict__ in_proj_b,
    const float* __restrict__ out_proj_w, const float* __restrict__ out_proj_b,
    const float* __restrict__ ln_g, const float* __restrict__ ln_b,
    const float* __restrict__ out_w, const float* __restrict__ out_b,
    const _Float16* __restrict__ wbuf,
    float* __restrict__ out, int n_tok)
{
    __shared__ SMem sm;
    const int tid  = threadIdx.x;
    const int lane = tid & 31;
    const int wave = tid >> 5;          // 0..2
    const int n0   = blockIdx.x * TOK;  // first token of this block
    const int mbase = wave * 16;

    // ---------- Stage 0: stage x tile into LDS as f16, pad K 45 -> 64 ----------
    for (int i = tid; i < ROWS * 64; i += BLOCK) {
        int r = i >> 6, c = i & 63;
        int tok = n0 + r / NA, a = r % NA;
        float v = 0.0f;
        if (c < BAND && tok < n_tok) {
            int kk = tok & (KSUB - 1);
            int bt = tok >> 3;
            v = x[(size_t)(bt * NA + a) * 360 + kk * BAND + c];
        }
        sm.X[r][c] = (_Float16)v;
    }
    __syncthreads();

    // ---------- Stage 1: h = X @ embed_w^T + embed_b ----------
    {
        v16h a0 = load_a_frag(&sm.X[0][0], mbase, 0, lane);
        v16h a1 = load_a_frag(&sm.X[0][0], mbase, 1, lane);
#pragma unroll
        for (int nt = 0; nt < 4; ++nt) {
            v8f acc = {};
            v16h b0, b1;
            if constexpr (W16F) {
                b0 = load_b_frag16(wbuf + EMB_OFF, nt, 0, lane);
                b1 = load_b_frag16(wbuf + EMB_OFF, nt, 1, lane);
            } else {
                b0 = load_b_frag32(embed_w, BAND, DM, nt, 0, lane);
                b1 = load_b_frag32(embed_w, BAND, DM, nt, 1, lane);
            }
            acc = wmma_f16(a0, b0, acc);
            acc = wmma_f16(a1, b1, acc);
            int col = nt * 16 + (lane & 15);
            float bias = embed_b[col];
#pragma unroll
            for (int r = 0; r < 8; ++r) {
                int row = mbase + r + 8 * (lane >> 4);
                float v = acc[r] + bias;
                sm.h32[row][col] = v;
                sm.h16[row][col] = (_Float16)v;
            }
        }
    }
    __syncthreads();

    // ---------- Stage 2: qkv = h @ in_proj_w^T + in_proj_b ----------
    {
        v16h a0 = load_a_frag(&sm.h16[0][0], mbase, 0, lane);
        v16h a1 = load_a_frag(&sm.h16[0][0], mbase, 1, lane);
#pragma unroll
        for (int nt = 0; nt < 12; ++nt) {
            v8f acc = {};
            v16h b0, b1;
            if constexpr (W16F) {
                b0 = load_b_frag16(wbuf + INP_OFF, nt, 0, lane);
                b1 = load_b_frag16(wbuf + INP_OFF, nt, 1, lane);
            } else {
                b0 = load_b_frag32(in_proj_w, DM, 3 * DM, nt, 0, lane);
                b1 = load_b_frag32(in_proj_w, DM, 3 * DM, nt, 1, lane);
            }
            acc = wmma_f16(a0, b0, acc);
            acc = wmma_f16(a1, b1, acc);
            int col = nt * 16 + (lane & 15);
            float bias = in_proj_b[col];
#pragma unroll
            for (int r = 0; r < 8; ++r) {
                int row = mbase + r + 8 * (lane >> 4);
                sm.qkv[row][col] = (_Float16)(acc[r] + bias);
            }
        }
    }
    __syncthreads();

    // ---------- Stage 3: per-token multi-head attention (VALU, f32) ----------
    for (int task = tid; task < TOK * NH * NA; task += BLOCK) {
        int t   = task / (NH * NA);
        int rem = task % (NH * NA);
        int hh  = rem / NA;
        int qa  = rem % NA;
        int rq  = t * NA + qa;

        float q[HD];
#pragma unroll
        for (int d = 0; d < HD; ++d) q[d] = (float)sm.qkv[rq][hh * HD + d];

        float s[NA];
        float mx = -3.0e38f;
#pragma unroll
        for (int b = 0; b < NA; ++b) {
            if (b == qa) { s[b] = 0.0f; continue; }
            float dot = 0.0f;
            int rk = t * NA + b;
#pragma unroll
            for (int d = 0; d < HD; ++d)
                dot += q[d] * (float)sm.qkv[rk][DM + hh * HD + d];
            s[b] = dot * 0.25f;          // 1/sqrt(16)
            mx = fmaxf(mx, s[b]);
        }
        float p[NA];
        float sum = 0.0f;
#pragma unroll
        for (int b = 0; b < NA; ++b) {
            p[b] = (b == qa) ? 0.0f : __expf(s[b] - mx);   // diagonal masked
            sum += p[b];
        }
        float inv = 1.0f / sum;
#pragma unroll
        for (int d = 0; d < HD; ++d) {
            float o = 0.0f;
#pragma unroll
            for (int b = 0; b < NA; ++b)
                o += p[b] * (float)sm.qkv[t * NA + b][2 * DM + hh * HD + d];
            sm.o16[rq][hh * HD + d] = (_Float16)(o * inv);
        }
    }
    __syncthreads();

    // ---------- Stage 4: out_proj + residual -> h32 (in place) ----------
    {
        v16h a0 = load_a_frag(&sm.o16[0][0], mbase, 0, lane);
        v16h a1 = load_a_frag(&sm.o16[0][0], mbase, 1, lane);
#pragma unroll
        for (int nt = 0; nt < 4; ++nt) {
            v8f acc = {};
            v16h b0, b1;
            if constexpr (W16F) {
                b0 = load_b_frag16(wbuf + OUTP_OFF, nt, 0, lane);
                b1 = load_b_frag16(wbuf + OUTP_OFF, nt, 1, lane);
            } else {
                b0 = load_b_frag32(out_proj_w, DM, DM, nt, 0, lane);
                b1 = load_b_frag32(out_proj_w, DM, DM, nt, 1, lane);
            }
            acc = wmma_f16(a0, b0, acc);
            acc = wmma_f16(a1, b1, acc);
            int col = nt * 16 + (lane & 15);
            float bias = out_proj_b[col];
#pragma unroll
            for (int r = 0; r < 8; ++r) {
                int row = mbase + r + 8 * (lane >> 4);
                sm.h32[row][col] = acc[r] + bias + sm.h32[row][col];
            }
        }
    }
    __syncthreads();

    // ---------- Stage 5: LayerNorm over 64 features -> y16 ----------
    if (tid < ROWS) {
        int row = tid;
        float mu = 0.0f;
#pragma unroll
        for (int c = 0; c < DM; ++c) mu += sm.h32[row][c];
        mu *= (1.0f / DM);
        float var = 0.0f;
#pragma unroll
        for (int c = 0; c < DM; ++c) {
            float d = sm.h32[row][c] - mu;
            var += d * d;
        }
        var *= (1.0f / DM);
        float inv = rsqrtf(var + 1e-5f);
#pragma unroll
        for (int c = 0; c < DM; ++c) {
            float yv = (sm.h32[row][c] - mu) * inv * ln_g[c] + ln_b[c];
            sm.y16[row][c] = (_Float16)yv;
        }
    }
    __syncthreads();

    // ---------- Stage 6: out = y @ out_w^T + out_b, scatter to global ----------
    {
        v16h a0 = load_a_frag(&sm.y16[0][0], mbase, 0, lane);
        v16h a1 = load_a_frag(&sm.y16[0][0], mbase, 1, lane);
#pragma unroll
        for (int nt = 0; nt < 3; ++nt) {
            v8f acc = {};
            v16h b0, b1;
            if constexpr (W16F) {
                b0 = load_b_frag16(wbuf + OUTW_OFF, nt, 0, lane);
                b1 = load_b_frag16(wbuf + OUTW_OFF, nt, 1, lane);
            } else {
                b0 = load_b_frag32(out_w, DM, BAND, nt, 0, lane);
                b1 = load_b_frag32(out_w, DM, BAND, nt, 1, lane);
            }
            acc = wmma_f16(a0, b0, acc);
            acc = wmma_f16(a1, b1, acc);
            int col = nt * 16 + (lane & 15);
            if (col < BAND) {
                float bias = out_b[col];
#pragma unroll
                for (int r = 0; r < 8; ++r) {
                    int row = mbase + r + 8 * (lane >> 4);
                    int tok = n0 + row / NA, a = row % NA;
                    if (tok < n_tok) {
                        int kk = tok & (KSUB - 1);
                        int bt = tok >> 3;
                        out[(size_t)(bt * NA + a) * 360 + kk * BAND + col] = acc[r] + bias;
                    }
                }
            }
        }
    }
}

extern "C" void kernel_launch(void* const* d_in, const int* in_sizes, int n_in,
                              void* d_out, int out_size, void* d_ws, size_t ws_size,
                              hipStream_t stream) {
    const float* x          = (const float*)d_in[0];
    const float* embed_w    = (const float*)d_in[1];
    const float* embed_b    = (const float*)d_in[2];
    const float* in_proj_w  = (const float*)d_in[3];
    const float* in_proj_b  = (const float*)d_in[4];
    const float* out_proj_w = (const float*)d_in[5];
    const float* out_proj_b = (const float*)d_in[6];
    const float* ln_g       = (const float*)d_in[7];
    const float* ln_b       = (const float*)d_in[8];
    const float* out_w      = (const float*)d_in[9];
    const float* out_b      = (const float*)d_in[10];
    float* out = (float*)d_out;

    int bt = in_sizes[0] / (NA * 360);
    int n_tok = bt * KSUB;
    int grid = (n_tok + TOK - 1) / TOK;

    if (ws_size >= (size_t)WBUF_ELEMS * sizeof(_Float16)) {
        _Float16* wbuf = (_Float16*)d_ws;
        convert_weights_kernel<<<(WBUF_ELEMS + 255) / 256, 256, 0, stream>>>(
            embed_w, in_proj_w, out_proj_w, out_w, wbuf);
        subcarrier_cross_attn_kernel<true><<<grid, BLOCK, 0, stream>>>(
            x, embed_w, embed_b, in_proj_w, in_proj_b, out_proj_w, out_proj_b,
            ln_g, ln_b, out_w, out_b, wbuf, out, n_tok);
    } else {
        subcarrier_cross_attn_kernel<false><<<grid, BLOCK, 0, stream>>>(
            x, embed_w, embed_b, in_proj_w, in_proj_b, out_proj_w, out_proj_b,
            ln_g, ln_b, out_w, out_b, nullptr, out, n_tok);
    }
}
